// DynamicMemory_57758720196681
// MI455X (gfx1250) — compile-verified, matrix operand order checked by reading, not run on hardware
//
#include <hip/hip_runtime.h>
#include <hip/hip_bf16.h>

// CDNA5 / gfx1250 WMMA types
typedef __attribute__((ext_vector_type(16))) __bf16 v16bf;
typedef __attribute__((ext_vector_type(8)))  float  v8f;

#define DDIM 256
#define TSTEPS 256
#define SSLOT 8
#define BT 4            // batches per workgroup
#define ROWS 32         // BT * SSLOT state rows per WG
#define CH 8            // time-chunk for hoisted x@W GEMM (CH*BT = 32 rows)

// ---------------------------------------------------------------------------
// Pre-kernel 1: keyV[s][e] = sum_d keys[s][d] * V[d][e]   (loop invariant)
// ---------------------------------------------------------------------------
__global__ void keyv_kernel(const float* __restrict__ keys,
                            const float* __restrict__ V,
                            float* __restrict__ keyv) {
  int s = blockIdx.x, e = threadIdx.x;
  float acc = 0.f;
  for (int d = 0; d < DDIM; ++d) acc += keys[s * DDIM + d] * V[d * DDIM + e];
  keyv[s * DDIM + e] = acc;
}

// ---------------------------------------------------------------------------
// Pre-kernel 2: Wt[n][k] = bf16(W[k][n])  (B-fragment friendly, in d_ws)
// ---------------------------------------------------------------------------
__global__ void wt_kernel(const float* __restrict__ W, __bf16* __restrict__ wt) {
  int idx = blockIdx.x * 256 + threadIdx.x;      // 32768 pairs
  int n = idx >> 7, k2 = (idx & 127) * 2;
  union { unsigned u; __bf16 h[2]; } pk;
  pk.h[0] = (__bf16)W[(size_t)k2 * DDIM + n];
  pk.h[1] = (__bf16)W[(size_t)(k2 + 1) * DDIM + n];
  *(unsigned*)&wt[n * DDIM + k2] = pk.u;
}

// Fragment K base for 16-bit A/B WMMA operands (ISA 7.12.2):
//   VGPR v, lane-half h: k = kt*32 + (v>=4?16:0) + h*8 + (v%4)*2
__device__ __forceinline__ int frag_kb(int kt, int v, int half) {
  return (kt << 5) + ((v & 4) << 2) + (half << 3) + ((v & 3) << 1);
}

// Load a v16bf A/B fragment from bf16 memory laid out K-contiguous:
// elements [base, base+8) -> vector elems 0..7, [base+16, base+24) -> 8..15
__device__ __forceinline__ v16bf load_frag(const __bf16* p) {
  union { v16bf v; uint4 q[2]; } u;
  u.q[0] = *(const uint4*)(p);
  u.q[1] = *(const uint4*)(p + 16);
  return u.v;
}

// ---------------------------------------------------------------------------
// Main recurrent kernel: one WG owns 4 batches (32 state rows) for all T.
// U's B-fragments live in registers for the whole time loop.
// ---------------------------------------------------------------------------
template <bool USE_WT>
__global__ __launch_bounds__(256, 1)
void dynmem_kernel(const float* __restrict__ inputs,    // [B,T,D]
                   const int*   __restrict__ lengths,   // [B]
                   const float* __restrict__ keys,      // [S,D]
                   const float* __restrict__ U,         // [D,D]
                   const float* __restrict__ W,         // [D,D] (f32 fallback)
                   const __bf16* __restrict__ wt,       // [D,D] bf16 transposed (ws)
                   const float* __restrict__ gate_bias, // [1]
                   const float* __restrict__ state_bias,// [D]
                   const float* __restrict__ keyv,      // [S,D] (ws)
                   float* __restrict__ out) {           // [B,S,D]
  // LDS-resident working set (~177 KB of the 320 KB/WGP)
  __shared__ float  sStates[ROWS * DDIM];  // carried states f32 (32 KB)
  __shared__ __bf16 sStatesB[ROWS * DDIM]; // bf16 shadow for WMMA A (16 KB)
  __shared__ float  sP[ROWS * DDIM];       // pre-activation / h buffer (32 KB)
  __shared__ float  sXW[ROWS * DDIM];      // hoisted x@W chunk (32 KB)
  __shared__ float  sXC[ROWS * DDIM];      // raw x chunk f32 (32 KB)
  __shared__ __bf16 sXCb[ROWS * DDIM];     // x chunk bf16 shadow (16 KB)
  __shared__ float  sKeys[SSLOT * DDIM];   // (8 KB)
  __shared__ float  sKeyV[SSLOT * DDIM];   // (8 KB)
  __shared__ float  sSB[DDIM];             // state_bias (1 KB)
  __shared__ float  sG[ROWS];              // gates

  const int tid = threadIdx.x;
  const int b0  = blockIdx.x * BT;

  const float gbias = gate_bias[0];
  const int len0 = lengths[b0 + 0];
  const int len1 = lengths[b0 + 1];
  const int len2 = lengths[b0 + 2];
  const int len3 = lengths[b0 + 3];

  // ---- one-time staging ----
  for (int i = tid; i < SSLOT * DDIM; i += 256) sKeys[i] = keys[i];
  for (int i = tid; i < SSLOT * DDIM; i += 256) sKeyV[i] = keyv[i];
  sSB[tid] = state_bias[tid];
  __syncthreads();
  // zero_state: states[b][s][:] = keys[s][:]  (both precisions)
  for (int i = tid; i < ROWS * DDIM; i += 256) {
    int r = i >> 8, d = i & 255;
    float kv = sKeys[(r & 7) * DDIM + d];
    sStates[i]  = kv;
    sStatesB[i] = (__bf16)kv;
  }

  const int wv   = tid >> 5;   // wave id 0..7
  const int ln   = tid & 31;
  const int nn   = ln & 15;    // lane's M/N index within a 16-tile
  const int half = ln >> 4;    // lane half (K-group select)
  const int cr   = tid >> 3;   // combine/gate row 0..31
  const int cq   = tid & 7;    // 8 lanes per row
  const int cb   = cr >> 3;    // local batch of this row
  const int cs   = cr & 7;     // slot of this row
  const int mylen = (cb == 0) ? len0 : (cb == 1) ? len1 : (cb == 2) ? len2 : len3;

  // ---- hoist U's B-fragments into registers for the entire T loop ----
  // wave wv owns output columns [wv*32, wv*32+32): 2 n-tiles x 8 k-tiles.
  v16bf bu[2][8];
#pragma unroll
  for (int nt = 0; nt < 2; ++nt) {
    const int col = (wv * 2 + nt) * 16 + nn;
#pragma unroll
    for (int kt = 0; kt < 8; ++kt) {
#pragma unroll
      for (int v = 0; v < 8; ++v) {
        int kb = frag_kb(kt, v, half);
        bu[nt][kt][2 * v]     = (__bf16)U[(size_t)kb * DDIM + col];
        bu[nt][kt][2 * v + 1] = (__bf16)U[(size_t)(kb + 1) * DDIM + col];
      }
    }
  }
  __syncthreads();

  for (int t = 0; t < TSTEPS; ++t) {
    const int st = t & (CH - 1);

    // ================= every 8 steps: stage x chunk, hoisted x@W GEMM ======
    if (st == 0) {
      for (int i = tid * 2; i < ROWS * DDIM; i += 512) {
        int row = i >> 8, d = i & 255;
        int stt = row >> 2, b = row & 3;          // row = stt*4 + b
        const float2 x2 = *(const float2*)&inputs[
            (((size_t)(b0 + b)) * TSTEPS + (t + stt)) * DDIM + d];
        *(float2*)&sXC[i] = x2;
        union { unsigned u; __bf16 h[2]; } pk;
        pk.h[0] = (__bf16)x2.x; pk.h[1] = (__bf16)x2.y;
        *(unsigned*)&sXCb[i] = pk.u;
      }
      __syncthreads();

      v8f acc[2][2] = {};
#pragma unroll
      for (int kt = 0; kt < 8; ++kt) {
        const int kbase = kt * 32 + half * 8;
        v16bf a[2];
#pragma unroll
        for (int mt = 0; mt < 2; ++mt)
          a[mt] = load_frag(&sXCb[(mt * 16 + nn) * DDIM + kbase]);
        v16bf bw[2];
#pragma unroll
        for (int nt = 0; nt < 2; ++nt) {
          const int col = (wv * 2 + nt) * 16 + nn;
          if (USE_WT) {
            bw[nt] = load_frag(&wt[col * DDIM + kbase]);
          } else {
#pragma unroll
            for (int v = 0; v < 8; ++v) {
              int kb = frag_kb(kt, v, half);
              bw[nt][2 * v]     = (__bf16)W[(size_t)kb * DDIM + col];
              bw[nt][2 * v + 1] = (__bf16)W[(size_t)(kb + 1) * DDIM + col];
            }
          }
        }
#pragma unroll
        for (int mt = 0; mt < 2; ++mt)
#pragma unroll
          for (int nt = 0; nt < 2; ++nt)
            acc[mt][nt] = __builtin_amdgcn_wmma_f32_16x16x32_bf16(
                false, a[mt], false, bw[nt], (short)0, acc[mt][nt], false, false);
      }
#pragma unroll
      for (int mt = 0; mt < 2; ++mt)
#pragma unroll
        for (int nt = 0; nt < 2; ++nt) {
          const int col = (wv * 2 + nt) * 16 + nn;
#pragma unroll
          for (int r8 = 0; r8 < 8; ++r8) {
            int row = mt * 16 + r8 + 8 * half;    // C layout: M = r8 + 8*half
            sXW[row * DDIM + col] = acc[mt][nt][r8];
          }
        }
      // prefetch next input chunk into caches (global_prefetch_b8)
      if (t + CH < TSTEPS) {
        const char* pf = (const char*)(inputs +
            (((size_t)(b0 + (tid >> 6))) * TSTEPS + (t + CH)) * DDIM);
        __builtin_prefetch(pf + (size_t)(tid & 63) * 128, 0, 1);
      }
      __syncthreads();
    }

    // ================= gate: g = sigmoid(<x, state+key> + bias) ===========
    {
      float acc = 0.f;
      const float* xrow = &sXC[(st * BT + cb) * DDIM];
#pragma unroll 4
      for (int i = 0; i < 32; ++i) {
        int d = cq * 32 + i;
        acc += xrow[d] * (sStates[cr * DDIM + d] + sKeys[cs * DDIM + d]);
      }
      acc += __shfl_xor(acc, 1);
      acc += __shfl_xor(acc, 2);
      acc += __shfl_xor(acc, 4);
      if (cq == 0) sG[cr] = 1.f / (1.f + __expf(-(acc + gbias)));
    }

    // ===== SU = states @ U : A from LDS bf16, B resident in registers =====
    {
      v8f acc[2][2] = {};
#pragma unroll
      for (int kt = 0; kt < 8; ++kt) {
        const int kbase = kt * 32 + half * 8;
        v16bf a[2];
#pragma unroll
        for (int mt = 0; mt < 2; ++mt)
          a[mt] = load_frag(&sStatesB[(mt * 16 + nn) * DDIM + kbase]);
#pragma unroll
        for (int mt = 0; mt < 2; ++mt)
#pragma unroll
          for (int nt = 0; nt < 2; ++nt)
            acc[mt][nt] = __builtin_amdgcn_wmma_f32_16x16x32_bf16(
                false, a[mt], false, bu[nt][kt], (short)0, acc[mt][nt], false, false);
      }
#pragma unroll
      for (int mt = 0; mt < 2; ++mt)
#pragma unroll
        for (int nt = 0; nt < 2; ++nt) {
          const int col = (wv * 2 + nt) * 16 + nn;
#pragma unroll
          for (int r8 = 0; r8 < 8; ++r8) {
            int row = mt * 16 + r8 + 8 * half;
            sP[row * DDIM + col] = acc[mt][nt][r8];
          }
        }
    }
    __syncthreads();

    // ======= combine: h = states + g*elu(SU + keyV + xW + bias); norm =====
    {
      const float g = sG[cr];
      const float* xwrow = &sXW[(st * BT + cb) * DDIM];
      float ss = 0.f;
#pragma unroll 4
      for (int i = 0; i < 32; ++i) {
        int d = cq * 32 + i;
        float pre = sP[cr * DDIM + d] + sKeyV[cs * DDIM + d] + xwrow[d] + sSB[d];
        float el  = pre > 0.f ? pre : (__expf(pre) - 1.f);
        float h   = sStates[cr * DDIM + d] + g * el;
        sP[cr * DDIM + d] = h;
        ss += h * h;
      }
      ss += __shfl_xor(ss, 1);
      ss += __shfl_xor(ss, 2);
      ss += __shfl_xor(ss, 4);
      float rn = 1.f / fmaxf(sqrtf(ss), 1e-12f);
      if (t < mylen) {                  // masked state update (both precisions)
#pragma unroll 4
        for (int i = 0; i < 32; i += 2) {
          int d = cq * 32 + i;
          float h0 = sP[cr * DDIM + d]     * rn;
          float h1 = sP[cr * DDIM + d + 1] * rn;
          sStates[cr * DDIM + d]     = h0;
          sStates[cr * DDIM + d + 1] = h1;
          union { unsigned u; __bf16 h[2]; } pk;
          pk.h[0] = (__bf16)h0; pk.h[1] = (__bf16)h1;
          *(unsigned*)&sStatesB[cr * DDIM + d] = pk.u;
        }
      }
    }
    __syncthreads();
  }

  // ---- writeback final states [B,S,D] ----
  for (int i = tid; i < ROWS * DDIM; i += 256) {
    int r = i >> 8, d = i & 255;
    int b = r >> 3, s = r & 7;
    out[(((size_t)(b0 + b)) * SSLOT + s) * DDIM + d] = sStates[i];
  }
}

// ---------------------------------------------------------------------------
extern "C" void kernel_launch(void* const* d_in, const int* in_sizes, int n_in,
                              void* d_out, int out_size, void* d_ws, size_t ws_size,
                              hipStream_t stream) {
  const float* inputs  = (const float*)d_in[0];
  const int*   lengths = (const int*)  d_in[1];
  const float* keys    = (const float*)d_in[2];
  const float* U       = (const float*)d_in[3];
  const float* V       = (const float*)d_in[4];
  const float* W       = (const float*)d_in[5];
  const float* gbias   = (const float*)d_in[6];
  const float* sbias   = (const float*)d_in[7];
  float* out  = (float*)d_out;

  float*  keyv = (float*)d_ws;                         // [0, 8KB)
  __bf16* wt   = (__bf16*)((char*)d_ws + 8192);        // [8KB, 8KB+128KB)
  const size_t need_wt = 8192 + (size_t)DDIM * DDIM * sizeof(__bf16);

  keyv_kernel<<<SSLOT, DDIM, 0, stream>>>(keys, V, keyv);
  if (ws_size >= need_wt) {
    wt_kernel<<<DDIM * DDIM / 512, 256, 0, stream>>>(W, wt);
    dynmem_kernel<true><<<512 / BT, 256, 0, stream>>>(
        inputs, lengths, keys, U, W, wt, gbias, sbias, keyv, out);
  } else {
    dynmem_kernel<false><<<512 / BT, 256, 0, stream>>>(
        inputs, lengths, keys, U, W, nullptr, gbias, sbias, keyv, out);
  }
}
